// LSTMDeepSimulator_64819646431333
// MI455X (gfx1250) — matrix operand order repaired and checked by reading, not compiled, not run on hardware
//
#include <hip/hip_runtime.h>

typedef __attribute__((ext_vector_type(16))) _Float16 v16h;
typedef __attribute__((ext_vector_type(8)))  float    v8f;
typedef __attribute__((ext_vector_type(4)))  float    v4f;

namespace {
constexpr int kB = 4096, kT = 2048, kH1 = 32, kH2 = 16;
constexpr int WAVES = 8;   // waves per block; each wave owns 16 batch rows
}

union V16HU { v4f f[2]; v16h v; };

__device__ __forceinline__ float fast_sigmoid(float x) {
  return __builtin_amdgcn_rcpf(1.0f + __expf(-x));
}
__device__ __forceinline__ float fast_tanh(float x) {
  float e = __expf(-2.0f * x);
  return (1.0f - e) * __builtin_amdgcn_rcpf(1.0f + e);
}

__device__ __forceinline__ v8f wmma_f16(v16h a, v16h b, v8f c) {
  // D = A(16x32 f16) * B(32x16 f16) + C(16x16 f32)
  return __builtin_amdgcn_wmma_f32_16x16x32_f16(false, a, false, b, (short)0, c,
                                                false, false);
}

// 16 consecutive f32 -> f16 packed into B/A-layout lane chunk (8 VGPRs)
__device__ __forceinline__ v16h pack16(const float* p) {
  v16h r;
#pragma unroll
  for (int i = 0; i < 16; ++i) r[i] = (_Float16)p[i];
  return r;
}

__global__ void __launch_bounds__(WAVES * 32)
lstm2_wmma_kernel(const float* __restrict__ x,
                  const float* __restrict__ W_ih1, const float* __restrict__ W_hh1,
                  const float* __restrict__ b_ih1, const float* __restrict__ b_hh1,
                  const float* __restrict__ W_ih2, const float* __restrict__ W_hh2,
                  const float* __restrict__ b_ih2, const float* __restrict__ b_hh2,
                  const float* __restrict__ W_out, const float* __restrict__ b_out,
                  float* __restrict__ out)
{
  // per-wave h tiles, row-major [batch-row][hidden] in f16 (for A-layout reload)
  __shared__ alignas(16) _Float16 h1buf[WAVES][16][32];
  __shared__ alignas(16) _Float16 h2buf[WAVES][16][32];  // cols 16..31 stay zero (K pad)
  // W_hh2^T in WMMA B-layout lane chunks, K padded 16->32 (shared by all waves)
  __shared__ alignas(16) _Float16 w2h[4][32][16];

  const int lane = threadIdx.x & 31;
  const int w    = threadIdx.x >> 5;
  const int n    = lane & 15;   // column within 16-wide N tile / batch row for A reads
  const int hi   = lane >> 4;   // which half of the wave
  const int b0   = (blockIdx.x * WAVES + w) * 16;
  const int kb   = hi * 16;     // B-layout K base for this lane

  // ---- loop-invariant weights in B-layout registers ----
  v16h WB1[8];                  // W_hh1^T tiles, N = 16t..16t+15, K = 0..31
#pragma unroll
  for (int t = 0; t < 8; ++t) WB1[t] = pack16(&W_hh1[(t * 16 + n) * kH1 + kb]);
  v16h WB2i[4];                 // W_ih2^T tiles
#pragma unroll
  for (int t = 0; t < 4; ++t) WB2i[t] = pack16(&W_ih2[(t * 16 + n) * kH1 + kb]);

  if (w == 0) {                 // zero-padded W_hh2^T tiles -> LDS (once)
#pragma unroll
    for (int t = 0; t < 4; ++t)
#pragma unroll
      for (int i = 0; i < 16; ++i)
        w2h[t][lane][i] = (_Float16)(hi == 0 ? W_hh2[(t * 16 + n) * kH2 + i] : 0.0f);
  }

  float B1v[8], wih1v[8];
#pragma unroll
  for (int t = 0; t < 8; ++t) {
    B1v[t]   = b_ih1[t * 16 + n] + b_hh1[t * 16 + n];
    wih1v[t] = W_ih1[t * 16 + n];        // N_IN == 1
  }
  float B2v[4];
#pragma unroll
  for (int t = 0; t < 4; ++t) B2v[t] = b_ih2[t * 16 + n] + b_hh2[t * 16 + n];
  const float woutv = W_out[n];
  const float bout  = b_out[0];

  {                                      // zero initial h state in LDS
    _Float16* p1 = &h1buf[w][0][0];
    _Float16* p2 = &h2buf[w][0][0];
#pragma unroll
    for (int i = lane; i < 16 * 32; i += 32) {
      p1[i] = (_Float16)0.f; p2[i] = (_Float16)0.f;
    }
  }
  __syncthreads();

  v8f  c1[2] = {};                       // c1 in D layout (j = ct*16 + n)
  v8f  c2    = {};
  v16h a1    = {};                       // h1_{t-1} in A layout (starts at 0)

  const float* xrow = x + (size_t)(b0 + n) * kT;

  for (int t = 0; t < kT; ++t) {
    // x_t for the 16 rows of this tile; broadcast to D-layout row pattern
    float xv = xrow[t];
    float xbc[8];
#pragma unroll
    for (int r = 0; r < 8; ++r) xbc[r] = __shfl(xv, r + hi * 8, 32);

    // ---------------- layer 1: gates = b + x*W_ih1 + h1 @ W_hh1^T ----------------
#pragma unroll
    for (int ct = 0; ct < 2; ++ct) {     // two 16-col halves of H1=32
      v8f g[4];                          // i, f, g, o tiles for this half
#pragma unroll
      for (int gate = 0; gate < 4; ++gate) {
        const int tt = gate * 2 + ct;    // gate tile index in 0..7
        v8f acc;
#pragma unroll
        for (int r = 0; r < 8; ++r) acc[r] = B1v[tt] + xbc[r] * wih1v[tt];
        g[gate] = wmma_f16(a1, WB1[tt], acc);
      }
#pragma unroll
      for (int r = 0; r < 8; ++r) {
        float c = fast_sigmoid(g[1][r]) * c1[ct][r] +
                  fast_sigmoid(g[0][r]) * fast_tanh(g[2][r]);
        c1[ct][r] = c;
        float h = fast_sigmoid(g[3][r]) * fast_tanh(c);
        h1buf[w][r + hi * 8][ct * 16 + n] = (_Float16)h;   // D -> row-major f16
      }
    }
    asm volatile("" ::: "memory");       // same-wave DS is in-order; stop IR reordering

    // reload h1_t in A layout; h2buf still holds h2_{t-1}
    v16h a2;
    {
      V16HU u;
      const v4f* p = (const v4f*)&h1buf[w][n][hi * 16];
      u.f[0] = p[0]; u.f[1] = p[1];      // K = kb..kb+15 contiguous
      a1 = u.v;
      const v4f* q = (const v4f*)&h2buf[w][n][hi * 16];
      u.f[0] = q[0]; u.f[1] = q[1];
      a2 = u.v;
    }

    // ------------- layer 2: gates = b + h1 @ W_ih2^T + h2 @ W_hh2^T -------------
    v8f g2[4];
#pragma unroll
    for (int tt = 0; tt < 4; ++tt) {
      v8f acc;
#pragma unroll
      for (int r = 0; r < 8; ++r) acc[r] = B2v[tt];
      V16HU u;
      const v4f* p = (const v4f*)&w2h[tt][lane][0];
      u.f[0] = p[0]; u.f[1] = p[1];
      acc    = wmma_f16(a1, WB2i[tt], acc);
      g2[tt] = wmma_f16(a2, u.v, acc);
    }

    v8f h2d;
#pragma unroll
    for (int r = 0; r < 8; ++r) {
      float c = fast_sigmoid(g2[1][r]) * c2[r] +
                fast_sigmoid(g2[0][r]) * fast_tanh(g2[2][r]);
      c2[r] = c;
      float h = fast_sigmoid(g2[3][r]) * fast_tanh(c);
      h2d[r] = h;
      h2buf[w][r + hi * 8][n] = (_Float16)h;   // cols 16..31 remain zero
    }
    asm volatile("" ::: "memory");

    // ---------------- output: y = h2 @ W_out^T + b_out ----------------
#pragma unroll
    for (int r = 0; r < 8; ++r) {
      float p = h2d[r] * woutv;
      p += __shfl_xor(p, 1, 32);
      p += __shfl_xor(p, 2, 32);
      p += __shfl_xor(p, 4, 32);
      p += __shfl_xor(p, 8, 32);         // all lanes of each half hold the dot
      if (n == r) out[(size_t)(b0 + r + hi * 8) * kT + t] = p + bout;
    }
  }
}

extern "C" void kernel_launch(void* const* d_in, const int* in_sizes, int n_in,
                              void* d_out, int out_size, void* d_ws, size_t ws_size,
                              hipStream_t stream) {
  (void)in_sizes; (void)n_in; (void)out_size; (void)d_ws; (void)ws_size;
  const float* x     = (const float*)d_in[0];
  const float* W_ih1 = (const float*)d_in[1];
  const float* W_hh1 = (const float*)d_in[2];
  const float* b_ih1 = (const float*)d_in[3];
  const float* b_hh1 = (const float*)d_in[4];
  const float* W_ih2 = (const float*)d_in[5];
  const float* W_hh2 = (const float*)d_in[6];
  const float* b_ih2 = (const float*)d_in[7];
  const float* b_hh2 = (const float*)d_in[8];
  const float* W_out = (const float*)d_in[9];
  const float* b_out = (const float*)d_in[10];
  float* out = (float*)d_out;

  const int blocks = kB / (16 * WAVES);  // 4096 rows / 128 rows per block = 32
  lstm2_wmma_kernel<<<dim3(blocks), dim3(WAVES * 32), 0, stream>>>(
      x, W_ih1, W_hh1, b_ih1, b_hh1, W_ih2, W_hh2, b_ih2, b_hh2, W_out, b_out, out);
}
